// Discriminator_3779571220680
// MI455X (gfx1250) — matrix operand order repaired and checked by reading, not compiled, not run on hardware
//
#include <hip/hip_runtime.h>
#include <math.h>

// ---------------- problem constants ----------------
#define Nn   256   // batch
#define Tt   256   // time steps
#define Hh   256   // hidden
#define G3   768   // 3*H
#define DIN  131   // GRU input width
#define DINP 160   // padded to 5 * 32
#define DDISC 292  // disc MLP input width
#define DDISCP 320 // padded to 10 * 32

typedef __attribute__((ext_vector_type(16))) _Float16 v16h;
typedef __attribute__((ext_vector_type(8)))  float    v8f;

#define WMMA_F16(A, B, C) \
  __builtin_amdgcn_wmma_f32_16x16x32_f16(false, (A), false, (B), (short)0, (C), false, false)

// first WMMA of an accumulation chain: C = inline zero
__device__ __forceinline__ v8f wmma_zero(v16h a, v16h b) {
  v8f z = {0.0f, 0.0f, 0.0f, 0.0f, 0.0f, 0.0f, 0.0f, 0.0f};
  return WMMA_F16(a, b, z);
}

// ---- A fragment (16x32 f16) from LDS, row-major, element stride `stride`.
// CDNA5 layout: lane<16 -> M=lane, K = {k0..k0+7, k0+16..k0+23};
//               lane>=16 -> M=lane-16, K = {k0+8..k0+15, k0+24..k0+31}.
__device__ __forceinline__ v16h load_a_lds(const _Float16* base, int stride, int k0, int lane) {
  const int m    = lane & 15;
  const int koff = (lane >> 4) * 8;
  const _Float16* p = base + (size_t)m * stride + k0 + koff;
  union { uint4 u; _Float16 h[8]; } lo, hi;
  lo.u = *(const uint4*)(p);
  hi.u = *(const uint4*)(p + 16);
  v16h a;
#pragma unroll
  for (int i = 0; i < 8; ++i) { a[i] = lo.h[i]; a[8 + i] = hi.h[i]; }
  return a;
}

// ---- B fragment (32x16 f16) from pre-converted f16 weights, row-major
// W[row][k] with k contiguous. Two 16B loads, no conversion, no guards.
__device__ __forceinline__ v16h load_b_f16(const _Float16* W, int ld, int row,
                                           int k0, int lane) {
  const _Float16* p = W + (size_t)row * ld + k0 + ((lane >> 4) << 3);
  union { uint4 u[2]; _Float16 h[16]; } q;
  q.u[0] = *(const uint4*)(p);
  q.u[1] = *(const uint4*)(p + 16);
  v16h b;
#pragma unroll
  for (int i = 0; i < 16; ++i) b[i] = q.h[i];
  return b;
}

__device__ __forceinline__ float sigmoid_f(float x) {
  return 1.0f / (1.0f + __expf(-x));
}
__device__ __forceinline__ float tanh_fast(float x) {
  return 1.0f - 2.0f / (__expf(2.0f * x) + 1.0f);   // saturates correctly
}

// =====================================================================
// Kernel 0: one-shot weight conversion to f16 (padded / transposed).
//   wih_h : [768][160]  (K padded 131->160 with zeros)
//   whh_h : [768][256]
//   w1t_h : [256][320]  (transposed W1, K padded 292->320 with zeros)
// =====================================================================
__global__ __launch_bounds__(256)
void k_prep(const float* __restrict__ W_ih, const float* __restrict__ W_hh,
            const float* __restrict__ W1,
            _Float16* __restrict__ wih_h, _Float16* __restrict__ whh_h,
            _Float16* __restrict__ w1t_h)
{
  const int stride = gridDim.x * blockDim.x;
  const int tid0 = blockIdx.x * blockDim.x + threadIdx.x;
  for (int i = tid0; i < G3 * DINP; i += stride) {
    const int row = i / DINP, k = i % DINP;
    wih_h[i] = (_Float16)((k < DIN) ? W_ih[row * DIN + k] : 0.0f);
  }
  for (int i = tid0; i < G3 * Hh; i += stride)
    whh_h[i] = (_Float16)W_hh[i];
  for (int i = tid0; i < Hh * DDISCP; i += stride) {
    const int row = i / DDISCP, k = i % DDISCP;   // row = output col of W1
    w1t_h[i] = (_Float16)((k < DDISC) ? W1[k * Hh + row] : 0.0f);
  }
}

// =====================================================================
// Kernel 1: embeddings + input projection.
// gx stored TRANSPOSED: gx[t][g3][n]  (n contiguous) so k_gru reads b128.
// grid (Tt, Nn/16), block 128 = 4 waves; A fragments register-resident,
// waves loop over 12 column groups of 64.
// =====================================================================
__global__ __launch_bounds__(128)
void k_gx(const float* __restrict__ et, const int* __restrict__ etype,
          const float* __restrict__ sl, const float* __restrict__ tpp,
          const float* __restrict__ wtab, const float* __restrict__ htab,
          const float* __restrict__ ttab,
          const _Float16* __restrict__ wih_h, const float* __restrict__ b_ih,
          _Float16* __restrict__ gx)
{
  const int t  = blockIdx.x;
  const int n0 = blockIdx.y * 16;
  const int tid  = threadIdx.x;
  const int lane = tid & 31;
  const int wave = tid >> 5;
  const int nl   = lane & 15;
  const int mbase = (lane >> 4) << 3;

  __shared__ __align__(16) _Float16 xs[16][DINP];

  // cooperative build of the 16 x 131 (padded 160) input tile
  for (int idx = tid; idx < 16 * DINP; idx += 128) {
    const int row = idx / DINP, k = idx % DINP;
    const size_t r = (size_t)(n0 + row) * Tt + t;
    float v = 0.0f;
    if (k == 0)        v = et[r];
    else if (k < 17)   { int wk = (int)fmodf(et[r], 7.0f);           v = wtab[wk * 16 + (k - 1)]; }
    else if (k < 33)   { int hr = (int)(fmodf(et[r], 1.0f) * 24.0f); v = htab[hr * 16 + (k - 17)]; }
    else if (k < 65)   v = ttab[etype[r] * 32 + (k - 33)];
    else if (k < 67)   v = sl[r * 2 + (k - 65)];
    else if (k < DIN)  v = tpp[r * 64 + (k - 67)];
    xs[row][k] = (_Float16)v;
  }
  __syncthreads();

  // A fragments once, reused for all 12 column groups
  v16h Af[5];
#pragma unroll
  for (int kc = 0; kc < 5; ++kc) Af[kc] = load_a_lds(&xs[0][0], DINP, kc * 32, lane);

  _Float16* gbase = gx + (size_t)t * G3 * Nn + n0 + mbase;

#pragma unroll
  for (int j = 0; j < 12; ++j) {
    const int col = j * 64 + wave * 16 + nl;
    const float bias = b_ih[col];
    v8f acc = wmma_zero(Af[0], load_b_f16(wih_h, DINP, col, 0, lane));
#pragma unroll
    for (int kc = 1; kc < 5; ++kc)
      acc = WMMA_F16(Af[kc], load_b_f16(wih_h, DINP, col, kc * 32, lane), acc);

    union { uint4 u; _Float16 h[8]; } o;
#pragma unroll
    for (int g = 0; g < 8; ++g) o.h[g] = (_Float16)(acc[g] + bias);
    *(uint4*)(gbase + (size_t)col * Nn) = o.u;       // one b128 per lane
  }
}

// =====================================================================
// Kernel 2: GRU recurrence. grid 16 WGs (16 independent sequences each),
// block 512 = 16 waves; wave w owns output columns [16w,16w+16) for all 3
// gates; W_hh slice register-resident (192 VGPRs of f16 B fragments).
// h double-buffered in LDS -> ONE barrier per step; h_old kept in f32 regs.
// Only 16 blocks exist (1/WGP) -> allow full register budget, no spills.
// =====================================================================
__global__ __launch_bounds__(512, 1)
void k_gru(const _Float16* __restrict__ gx, const _Float16* __restrict__ whh_h,
           const float* __restrict__ b_hh, _Float16* __restrict__ hs)
{
  const int n0   = blockIdx.x * 16;
  const int tid  = threadIdx.x;
  const int lane = tid & 31;
  const int wave = tid >> 5;
  const int c0   = wave * 16;
  const int nl   = lane & 15;
  const int mbase = (lane >> 4) << 3;

  __shared__ __align__(16) _Float16 hbuf[2][16][Hh];   // 16 KB, double buffered

  for (int idx = tid; idx < 2 * 16 * Hh; idx += 512)
    (&hbuf[0][0][0])[idx] = (_Float16)0.0f;            // h0 = 0

  // register-resident W_hh fragments: 3 gates x 8 k-chunks x 8 VGPRs
  v16h Bw[3][8];
#pragma unroll
  for (int gate = 0; gate < 3; ++gate)
#pragma unroll
    for (int kc = 0; kc < 8; ++kc)
      Bw[gate][kc] = load_b_f16(whh_h, Hh, gate * Hh + c0 + nl, kc * 32, lane);

  const float bhr = b_hh[c0 + nl];
  const float bhz = b_hh[Hh + c0 + nl];
  const float bhn = b_hh[2 * Hh + c0 + nl];

  float hprev[8];
#pragma unroll
  for (int g = 0; g < 8; ++g) hprev[g] = 0.0f;

  // copy-duty indices for the vectorized hs writeback
  const int crow = tid >> 5;            // 0..15
  const int ccol = (tid & 31) * 8;      // 0..248
  _Float16* hsrow = hs + ((size_t)(n0 + crow) * Tt) * Hh + ccol;

  __syncthreads();

  for (int t = 0; t < Tt; ++t) {
    const int cur = t & 1, nxt = cur ^ 1;
    const _Float16* gx_t = gx + (size_t)t * G3 * Nn + n0 + mbase;

    if (t + 1 < Tt) {                    // prefetch next step's gate slices
      const _Float16* gx_n = gx_t + (size_t)G3 * Nn;
      __builtin_prefetch(gx_n + (size_t)(c0 + nl) * Nn, 0, 1);
      __builtin_prefetch(gx_n + (size_t)(Hh + c0 + nl) * Nn, 0, 1);
      __builtin_prefetch(gx_n + (size_t)(2 * Hh + c0 + nl) * Nn, 0, 1);
    }

    // gate preactivations from gx: 3 x b128 per lane (n contiguous)
    union { uint4 u; _Float16 h[8]; } gr, gz, gn;
    gr.u = *(const uint4*)(gx_t + (size_t)(c0 + nl) * Nn);
    gz.u = *(const uint4*)(gx_t + (size_t)(Hh + c0 + nl) * Nn);
    gn.u = *(const uint4*)(gx_t + (size_t)(2 * Hh + c0 + nl) * Nn);

    // zero-C chain start (inline 0 SRC2), bias folded into the tail
    v16h a0 = load_a_lds(&hbuf[cur][0][0], Hh, 0, lane);
    v8f ar = wmma_zero(a0, Bw[0][0]);
    v8f az = wmma_zero(a0, Bw[1][0]);
    v8f an = wmma_zero(a0, Bw[2][0]);
#pragma unroll
    for (int kc = 1; kc < 8; ++kc) {
      v16h a = load_a_lds(&hbuf[cur][0][0], Hh, kc * 32, lane);
      ar = WMMA_F16(a, Bw[0][kc], ar);
      az = WMMA_F16(a, Bw[1][kc], az);
      an = WMMA_F16(a, Bw[2][kc], an);
    }

#pragma unroll
    for (int g = 0; g < 8; ++g) {
      const float r  = sigmoid_f((float)gr.h[g] + bhr + ar[g]);
      const float z  = sigmoid_f((float)gz.h[g] + bhz + az[g]);
      const float nn = tanh_fast((float)gn.h[g] + r * (bhn + an[g])); // gh_n (incl. b_hh) kept separate
      hprev[g] = (1.0f - z) * nn + z * hprev[g];            // h_old from registers
      hbuf[nxt][mbase + g][c0 + nl] = (_Float16)hprev[g];
    }
    __syncthreads();                                        // h_{t+1} visible

    // vectorized writeback of h_{t+1}: one b128 store per thread
    *(uint4*)(hsrow + (size_t)t * Hh) = *(const uint4*)(&hbuf[nxt][crow][ccol]);
  }
}

// =====================================================================
// Kernel 3: disc MLP. grid N*T/16, block 512 = 16 waves. Layer 1 via WMMA
// (K=292 padded to 320, f16 transposed W1), tanh to LDS, layer 2 as
// per-wave dot + shfl reduce.
// =====================================================================
__global__ __launch_bounds__(512)
void k_disc(const _Float16* __restrict__ hs, const float* __restrict__ inter,
            const int* __restrict__ tact, const float* __restrict__ lact,
            const float* __restrict__ sl, const float* __restrict__ ttab,
            const _Float16* __restrict__ w1t_h, const float* __restrict__ b1,
            const float* __restrict__ W2, const float* __restrict__ b2,
            float* __restrict__ out)
{
  const int r0   = blockIdx.x * 16;
  const int tid  = threadIdx.x;
  const int lane = tid & 31;
  const int wave = tid >> 5;
  const int nl   = lane & 15;
  const int mbase = (lane >> 4) << 3;

  __shared__ __align__(16) _Float16 fs[16][DDISCP];  // 10 KB
  __shared__ __align__(16) _Float16 h1s[16][Hh];     // 8 KB

  // part 1: vector copy of h (16 rows x 256) -- one b128 per thread
  {
    const int row = tid >> 5, c = (tid & 31) * 8;
    *(uint4*)(&fs[row][c]) =
        *(const uint4*)(hs + (size_t)(r0 + row) * Hh + c);
  }
  // part 2: extras, k in [256, 320)
  for (int idx = tid; idx < 16 * 64; idx += 512) {
    const int row = idx >> 6, k = 256 + (idx & 63);
    const size_t r = (size_t)(r0 + row);
    float v = 0.0f;
    if (k == 256)        v = inter[r];
    else if (k < 289)    v = ttab[tact[r] * 32 + (k - 257)];
    else if (k < 291)    v = lact[r * 2 + (k - 289)];
    else if (k == 291) {
      const float dx = sl[r * 2]     - lact[r * 2];
      const float dy = sl[r * 2 + 1] - lact[r * 2 + 1];
      v = sqrtf(dx * dx + dy * dy);
    }
    fs[row][k] = (_Float16)v;
  }
  __syncthreads();

  const int col = wave * 16 + nl;
  const float bias = b1[col];
  v8f acc = wmma_zero(load_a_lds(&fs[0][0], DDISCP, 0, lane),
                      load_b_f16(w1t_h, DDISCP, col, 0, lane));
#pragma unroll
  for (int kc = 1; kc < 10; ++kc) {
    v16h a = load_a_lds(&fs[0][0], DDISCP, kc * 32, lane);
    acc = WMMA_F16(a, load_b_f16(w1t_h, DDISCP, col, kc * 32, lane), acc);
  }
#pragma unroll
  for (int g = 0; g < 8; ++g)
    h1s[mbase + g][col] = (_Float16)tanh_fast(acc[g] + bias);
  __syncthreads();

  // layer 2: wave w reduces row w over 256 columns
  float partial = 0.0f;
  const _Float16* hrow = &h1s[wave][0];
#pragma unroll
  for (int j = 0; j < 8; ++j) {
    const int k = lane * 8 + j;
    partial += (float)hrow[k] * W2[k];
  }
#pragma unroll
  for (int off = 16; off > 0; off >>= 1) partial += __shfl_xor(partial, off, 32);
  if (lane == 0) out[r0 + wave] = sigmoid_f(partial + b2[0]);
}

// =====================================================================
extern "C" void kernel_launch(void* const* d_in, const int* in_sizes, int n_in,
                              void* d_out, int out_size, void* d_ws, size_t ws_size,
                              hipStream_t stream) {
  (void)in_sizes; (void)n_in; (void)out_size; (void)ws_size;
  const float* event_times = (const float*)d_in[0];
  const int*   event_types = (const int*)  d_in[1];
  const float* sl          = (const float*)d_in[2];
  const float* tpp         = (const float*)d_in[3];
  const float* inter       = (const float*)d_in[4];
  const int*   tact        = (const int*)  d_in[5];
  const float* lact        = (const float*)d_in[6];
  const float* wtab        = (const float*)d_in[7];
  const float* htab        = (const float*)d_in[8];
  const float* ttab        = (const float*)d_in[9];
  const float* W_ih        = (const float*)d_in[10];
  const float* W_hh        = (const float*)d_in[11];
  const float* b_ih        = (const float*)d_in[12];
  const float* b_hh        = (const float*)d_in[13];
  const float* W1          = (const float*)d_in[14];
  const float* b1          = (const float*)d_in[15];
  const float* W2          = (const float*)d_in[16];
  const float* b2          = (const float*)d_in[17];

  char* ws = (char*)d_ws;
  const size_t gx_bytes  = (size_t)Tt * G3 * Nn * 2;   // 96 MiB, layout [t][g3][n]
  const size_t hs_bytes  = (size_t)Nn * Tt * Hh * 2;   // 32 MiB, layout [n][t][h]
  const size_t wih_bytes = (size_t)G3 * DINP * 2;
  const size_t whh_bytes = (size_t)G3 * Hh * 2;

  _Float16* gx    = (_Float16*)(ws);
  _Float16* hsb   = (_Float16*)(ws + gx_bytes);
  _Float16* wih_h = (_Float16*)(ws + gx_bytes + hs_bytes);
  _Float16* whh_h = (_Float16*)(ws + gx_bytes + hs_bytes + wih_bytes);
  _Float16* w1t_h = (_Float16*)(ws + gx_bytes + hs_bytes + wih_bytes + whh_bytes);

  k_prep<<<dim3(768), 256, 0, stream>>>(W_ih, W_hh, W1, wih_h, whh_h, w1t_h);
  k_gx  <<<dim3(Tt, Nn / 16), 128, 0, stream>>>(
      event_times, event_types, sl, tpp, wtab, htab, ttab, wih_h, b_ih, gx);
  k_gru <<<dim3(Nn / 16), 512, 0, stream>>>(gx, whh_h, b_hh, hsb);
  k_disc<<<dim3((Nn * Tt) / 16), 512, 0, stream>>>(
      hsb, inter, tact, lact, sl, ttab, w1t_h, b1, W2, b2, (float*)d_out);
}
